// MutilQueryAttention_64484638982857
// MI455X (gfx1250) — matrix operand order repaired and checked by reading, not compiled, not run on hardware
//
#include <hip/hip_runtime.h>

// ---------------------------------------------------------------------------
// MQA forward for gfx1250 (CDNA5, wave32, WMMA bf16 16x16x32 + async->LDS)
// B=2, S=2048, D=1024, H=16, HD=64, SCALE = 1/8
// ---------------------------------------------------------------------------

typedef __attribute__((ext_vector_type(16))) __bf16 v16bf;
typedef __attribute__((ext_vector_type(8)))  __bf16 v8bf;
typedef __attribute__((ext_vector_type(8)))  float  v8f;

#define MQA_B   2
#define MQA_S   2048
#define MQA_D   1024
#define MQA_H   16
#define MQA_HD  64
#define MQA_M   (MQA_B * MQA_S)      // 4096 tokens
#define MQA_SCALE 0.125f

__device__ __forceinline__ v16bf cat8(v8bf lo, v8bf hi) {
  return __builtin_shufflevector(lo, hi, 0,1,2,3,4,5,6,7,8,9,10,11,12,13,14,15);
}

// 16 contiguous bf16 starting at p (per-lane pointer; global or LDS)
__device__ __forceinline__ v16bf ld16(const __bf16* p) {
  return cat8(*(const v8bf*)(p), *(const v8bf*)(p + 8));
}

__device__ __forceinline__ v8f wmma_bf16(v16bf a, v16bf b, v8f c) {
  // (neg_a, A, neg_b, B, c_mod, C, reuse_a, reuse_b)
  return __builtin_amdgcn_wmma_f32_16x16x32_bf16(false, a, false, b, (short)0, c,
                                                 false, false);
}

// Async copy 16 bytes global -> LDS (ASYNCcnt-tracked), GV addressing.
__device__ __forceinline__ void async_b128(unsigned lds_addr, const void* gptr) {
  asm volatile("global_load_async_to_lds_b128 %0, %1, off"
               :: "v"(lds_addr), "v"((unsigned long long)(uintptr_t)gptr)
               : "memory");
}
__device__ __forceinline__ void wait_asynccnt0() {
  asm volatile("s_wait_asynccnt 0x0" ::: "memory");
}

// ---------------------------------------------------------------------------
// Kernel 0a: elementwise fp32 -> bf16
// ---------------------------------------------------------------------------
__global__ void cvt_bf16_kernel(const float* __restrict__ src,
                                __bf16* __restrict__ dst, int n) {
  int i = blockIdx.x * blockDim.x + threadIdx.x;
  if (i < n) dst[i] = (__bf16)src[i];
}

// ---------------------------------------------------------------------------
// Kernel 0b: transpose + convert.  src [R,C] row-major fp32 -> dst [C,R] bf16
// ---------------------------------------------------------------------------
__global__ void transpose_cvt_kernel(const float* __restrict__ src,
                                     __bf16* __restrict__ dst, int R, int C) {
  int i = blockIdx.x * blockDim.x + threadIdx.x;
  if (i < R * C) {
    int r = i % R;            // dst fast dim (contiguous k)
    int c = i / R;
    dst[i] = (__bf16)src[(size_t)r * C + c];
  }
}

// ---------------------------------------------------------------------------
// Kernel 1: fused QKV projection.  One wave per 32x64 C-tile (2x4 WMMA tiles).
//   grid.x : token tile (M/32 = 128)
//   grid.y : 0..15 -> Q (64-col blocks), 16 -> K (64 cols), 17 -> V (64 cols)
// Outputs: Qbf [M,1024] row-major, Kbf [M,64] row-major, Vt [64, M] (transposed)
// ---------------------------------------------------------------------------
__global__ void __launch_bounds__(32)
qkv_gemm_kernel(const __bf16* __restrict__ X,
                const __bf16* __restrict__ WqT,
                const __bf16* __restrict__ WkT,
                const __bf16* __restrict__ WvT,
                __bf16* __restrict__ Qbf,
                __bf16* __restrict__ Kbf,
                __bf16* __restrict__ Vt) {
  const int lane = threadIdx.x;
  const int col  = lane & 15;
  const int half = lane >> 4;
  const int m0   = blockIdx.x * 32;
  const int nt   = blockIdx.y;

  const __bf16* WT;
  int n0, mode;
  if (nt < 16)      { WT = WqT; n0 = nt * 64; mode = 0; }
  else if (nt == 16){ WT = WkT; n0 = 0;       mode = 1; }
  else              { WT = WvT; n0 = 0;       mode = 2; }

  const __bf16* arow0 = X + (size_t)(m0 + col) * MQA_D;
  const __bf16* arow1 = X + (size_t)(m0 + 16 + col) * MQA_D;
  const __bf16* brow0 = WT + (size_t)(n0 + col) * MQA_D + half * 16;
  const __bf16* brow1 = brow0 + (size_t)16 * MQA_D;
  const __bf16* brow2 = brow0 + (size_t)32 * MQA_D;
  const __bf16* brow3 = brow0 + (size_t)48 * MQA_D;

  v8f acc[2][4] = {};
  for (int kc = 0; kc < MQA_D; kc += 32) {
    const v16bf a0 = cat8(*(const v8bf*)(arow0 + kc + half * 16),
                          *(const v8bf*)(arow0 + kc + 8 + half * 16));
    const v16bf a1 = cat8(*(const v8bf*)(arow1 + kc + half * 16),
                          *(const v8bf*)(arow1 + kc + 8 + half * 16));
    const v16bf b0 = ld16(brow0 + kc);
    acc[0][0] = wmma_bf16(a0, b0, acc[0][0]);
    acc[1][0] = wmma_bf16(a1, b0, acc[1][0]);
    const v16bf b1 = ld16(brow1 + kc);
    acc[0][1] = wmma_bf16(a0, b1, acc[0][1]);
    acc[1][1] = wmma_bf16(a1, b1, acc[1][1]);
    const v16bf b2 = ld16(brow2 + kc);
    acc[0][2] = wmma_bf16(a0, b2, acc[0][2]);
    acc[1][2] = wmma_bf16(a1, b2, acc[1][2]);
    const v16bf b3 = ld16(brow3 + kc);
    acc[0][3] = wmma_bf16(a0, b3, acc[0][3]);
    acc[1][3] = wmma_bf16(a1, b3, acc[1][3]);
  }

#pragma unroll
  for (int mi = 0; mi < 2; ++mi) {
#pragma unroll
    for (int j = 0; j < 4; ++j) {
#pragma unroll
      for (int r = 0; r < 8; ++r) {
        const int row   = mi * 16 + r + 8 * half;
        const int token = m0 + row;
        const int n     = n0 + j * 16 + col;
        const __bf16 v  = (__bf16)acc[mi][j][r];
        if (mode == 0)      Qbf[(size_t)token * MQA_D + n]  = v;
        else if (mode == 1) Kbf[(size_t)token * MQA_HD + n] = v;
        else                Vt[(size_t)n * MQA_M + token]   = v;  // transposed
      }
    }
  }
}

// ---------------------------------------------------------------------------
// Kernel 2: flash-attention MQA.
//   Block = 128 threads (4 waves); each wave owns one 16-query tile.
//   Wave 0 stages the shared 32-key K and V chunks into LDS with
//   global_load_async_to_lds_b128 (ASYNCcnt), barrier releases all waves.
//   grid = (S/64, H, B)
// ---------------------------------------------------------------------------
__global__ void __launch_bounds__(128)
mqa_flash_kernel(const __bf16* __restrict__ Qbf,
                 const __bf16* __restrict__ Kbf,
                 const __bf16* __restrict__ Vt,
                 __bf16* __restrict__ Ctx) {
  __shared__ __bf16 Klds[32 * MQA_HD];       // [key][hd]   4 KB
  __shared__ __bf16 Vlds[MQA_HD * 32];       // [hd][key]   4 KB
  __shared__ __bf16 Plds[4][16 * 32];        // per-wave P bounce, 4 KB

  const int tid  = threadIdx.x;
  const int wv   = tid >> 5;
  const int lane = tid & 31;
  const int col  = lane & 15;
  const int half = lane >> 4;
  const int q0   = blockIdx.x * 64 + wv * 16;
  const int h    = blockIdx.y;
  const int b    = blockIdx.z;

  // --- Q tile in registers (A layout), hd split into two K=32 steps ---------
  const __bf16* qrow = Qbf + (size_t)(b * MQA_S + q0 + col) * MQA_D + h * MQA_HD;
  const v16bf qa0 = cat8(*(const v8bf*)(qrow +      half * 16),
                         *(const v8bf*)(qrow + 8 + half * 16));
  const v16bf qa1 = cat8(*(const v8bf*)(qrow + 32 + half * 16),
                         *(const v8bf*)(qrow + 40 + half * 16));

  v8f acc0 = {}, acc1 = {}, acc2 = {}, acc3 = {};
  float mrow[8], lrow[8];
#pragma unroll
  for (int r = 0; r < 8; ++r) { mrow[r] = -3.0e38f; lrow[r] = 0.0f; }

  const __bf16* Kb = Kbf + (size_t)b * MQA_S * MQA_HD;
  const __bf16* Vb = Vt + (size_t)b * MQA_S;          // hd-row stride = MQA_M
  __bf16* myP = &Plds[wv][0];

  for (int kc = 0; kc < MQA_S; kc += 32) {
    __syncthreads();                         // prior chunk fully consumed
    if (wv == 0) {
      // K chunk: 32 keys x 64 hd = 4 KB, contiguous in Kbf.
      const char* kbase = (const char*)(Kb + (size_t)kc * MQA_HD);
      const unsigned klds = (unsigned)(uintptr_t)&Klds[0];
#pragma unroll
      for (int it = 0; it < 8; ++it) {
        const int id = it * 32 + lane;       // 16B chunk index
        async_b128(klds + id * 16, kbase + id * 16);
      }
      // V chunk: 64 hd-rows x 32 keys -> 64 segments of 64 B.
      const char* vbase = (const char*)Vb + (size_t)kc * 2;
      const unsigned vlds = (unsigned)(uintptr_t)&Vlds[0];
#pragma unroll
      for (int it = 0; it < 8; ++it) {
        const int id  = it * 32 + lane;      // 16B chunk index (256 total)
        const int hd  = id >> 2;
        const int off = (id & 3) * 16;
        async_b128(vlds + hd * 64 + off,
                   vbase + (size_t)hd * (MQA_M * 2) + off);
      }
      wait_asynccnt0();
    } else if (wv == 1 && kc + 32 < MQA_S) { // prefetch next K chunk to L2
      __builtin_prefetch((const char*)(Kb + (size_t)(kc + 32) * MQA_HD) +
                             lane * 128, 0, 1);
    }
    __syncthreads();                         // K/V chunk visible in LDS

    // ---- scores: S = Q @ K^T from LDS (two 16-key N tiles) -----------------
    const __bf16* kr0 = &Klds[(size_t)col * MQA_HD + half * 16];
    const __bf16* kr1 = &Klds[(size_t)(16 + col) * MQA_HD + half * 16];
    v8f s0 = {}, s1 = {};
    s0 = wmma_bf16(qa0, ld16(kr0),      s0);
    s0 = wmma_bf16(qa1, ld16(kr0 + 32), s0);
    s1 = wmma_bf16(qa0, ld16(kr1),      s1);
    s1 = wmma_bf16(qa1, ld16(kr1 + 32), s1);

    // ---- online softmax (row stats per C-layout row r + 8*half) ------------
    float p0[8], p1[8], alpha[8];
#pragma unroll
    for (int r = 0; r < 8; ++r) {
      float t0 = s0[r] * MQA_SCALE;
      float t1 = s1[r] * MQA_SCALE;
      float mx = fmaxf(t0, t1);
      mx = fmaxf(mx, __shfl_xor(mx, 1, 32));
      mx = fmaxf(mx, __shfl_xor(mx, 2, 32));
      mx = fmaxf(mx, __shfl_xor(mx, 4, 32));
      mx = fmaxf(mx, __shfl_xor(mx, 8, 32));   // row max within 16-lane half
      const float mnew = fmaxf(mrow[r], mx);
      const float a    = __expf(mrow[r] - mnew);
      const float e0   = __expf(t0 - mnew);
      const float e1   = __expf(t1 - mnew);
      float sum = e0 + e1;
      sum += __shfl_xor(sum, 1, 32);
      sum += __shfl_xor(sum, 2, 32);
      sum += __shfl_xor(sum, 4, 32);
      sum += __shfl_xor(sum, 8, 32);
      lrow[r]  = lrow[r] * a + sum;
      mrow[r]  = mnew;
      alpha[r] = a;
      p0[r] = e0;
      p1[r] = e1;
    }
#pragma unroll
    for (int r = 0; r < 8; ++r) {
      acc0[r] *= alpha[r]; acc1[r] *= alpha[r];
      acc2[r] *= alpha[r]; acc3[r] *= alpha[r];
    }

    // ---- P: C-layout -> A-layout via private LDS region --------------------
#pragma unroll
    for (int r = 0; r < 8; ++r) {
      const int row = r + 8 * half;
      myP[row * 32 + col]      = (__bf16)p0[r];
      myP[row * 32 + 16 + col] = (__bf16)p1[r];
    }
    __syncthreads();   // uniform; also orders per-wave P store -> load
    const v16bf pa = cat8(*(const v8bf*)(&myP[col * 32 + half * 16]),
                          *(const v8bf*)(&myP[col * 32 + 8 + half * 16]));

    // ---- ctx += P @ V from LDS ---------------------------------------------
    const __bf16* vp = &Vlds[(size_t)col * 32 + half * 16];
    acc0 = wmma_bf16(pa, ld16(vp),            acc0);
    acc1 = wmma_bf16(pa, ld16(vp + 16 * 32),  acc1);
    acc2 = wmma_bf16(pa, ld16(vp + 32 * 32),  acc2);
    acc3 = wmma_bf16(pa, ld16(vp + 48 * 32),  acc3);
  }

  // ---- normalize and store ctx tile (bf16) ---------------------------------
#pragma unroll
  for (int r = 0; r < 8; ++r) {
    const int row  = r + 8 * half;
    const float iv = 1.0f / lrow[r];
    const size_t base =
        (size_t)(b * MQA_S + q0 + row) * MQA_D + h * MQA_HD + col;
    Ctx[base]      = (__bf16)(acc0[r] * iv);
    Ctx[base + 16] = (__bf16)(acc1[r] * iv);
    Ctx[base + 32] = (__bf16)(acc2[r] * iv);
    Ctx[base + 48] = (__bf16)(acc3[r] * iv);
  }
}

// ---------------------------------------------------------------------------
// Kernel 3: output projection Out = Ctx @ Wo (fp32 out). 32x64 tile per wave.
// ---------------------------------------------------------------------------
__global__ void __launch_bounds__(32)
out_gemm_kernel(const __bf16* __restrict__ Ctx,
                const __bf16* __restrict__ WoT,
                float* __restrict__ Out) {
  const int lane = threadIdx.x;
  const int col  = lane & 15;
  const int half = lane >> 4;
  const int m0   = blockIdx.x * 32;
  const int n0   = blockIdx.y * 64;

  const __bf16* arow0 = Ctx + (size_t)(m0 + col) * MQA_D;
  const __bf16* arow1 = Ctx + (size_t)(m0 + 16 + col) * MQA_D;
  const __bf16* brow0 = WoT + (size_t)(n0 + col) * MQA_D + half * 16;
  const __bf16* brow1 = brow0 + (size_t)16 * MQA_D;
  const __bf16* brow2 = brow0 + (size_t)32 * MQA_D;
  const __bf16* brow3 = brow0 + (size_t)48 * MQA_D;

  v8f acc[2][4] = {};
  for (int kc = 0; kc < MQA_D; kc += 32) {
    const v16bf a0 = cat8(*(const v8bf*)(arow0 + kc + half * 16),
                          *(const v8bf*)(arow0 + kc + 8 + half * 16));
    const v16bf a1 = cat8(*(const v8bf*)(arow1 + kc + half * 16),
                          *(const v8bf*)(arow1 + kc + 8 + half * 16));
    const v16bf b0 = ld16(brow0 + kc);
    acc[0][0] = wmma_bf16(a0, b0, acc[0][0]);
    acc[1][0] = wmma_bf16(a1, b0, acc[1][0]);
    const v16bf b1 = ld16(brow1 + kc);
    acc[0][1] = wmma_bf16(a0, b1, acc[0][1]);
    acc[1][1] = wmma_bf16(a1, b1, acc[1][1]);
    const v16bf b2 = ld16(brow2 + kc);
    acc[0][2] = wmma_bf16(a0, b2, acc[0][2]);
    acc[1][2] = wmma_bf16(a1, b2, acc[1][2]);
    const v16bf b3 = ld16(brow3 + kc);
    acc[0][3] = wmma_bf16(a0, b3, acc[0][3]);
    acc[1][3] = wmma_bf16(a1, b3, acc[1][3]);
  }

#pragma unroll
  for (int mi = 0; mi < 2; ++mi) {
#pragma unroll
    for (int j = 0; j < 4; ++j) {
#pragma unroll
      for (int r = 0; r < 8; ++r) {
        const int row = mi * 16 + r + 8 * half;
        Out[(size_t)(m0 + row) * MQA_D + n0 + j * 16 + col] = acc[mi][j][r];
      }
    }
  }
}

// ---------------------------------------------------------------------------
// Host launcher
// ---------------------------------------------------------------------------
extern "C" void kernel_launch(void* const* d_in, const int* in_sizes, int n_in,
                              void* d_out, int out_size, void* d_ws, size_t ws_size,
                              hipStream_t stream) {
  const float* Qin = (const float*)d_in[0];   // [B,S,D]
  const float* Wq  = (const float*)d_in[1];   // [D,1024]
  const float* Wk  = (const float*)d_in[2];   // [D,64]
  const float* Wv  = (const float*)d_in[3];   // [D,64]
  const float* Wo  = (const float*)d_in[4];   // [1024,D]
  float* Out = (float*)d_out;

  char* p = (char*)d_ws;
  __bf16* Xbf = (__bf16*)p; p += (size_t)MQA_M * MQA_D * 2;      // 8 MB
  __bf16* WqT = (__bf16*)p; p += (size_t)MQA_D * MQA_D * 2;      // 2 MB
  __bf16* WkT = (__bf16*)p; p += (size_t)MQA_HD * MQA_D * 2;     // 128 KB
  __bf16* WvT = (__bf16*)p; p += (size_t)MQA_HD * MQA_D * 2;     // 128 KB
  __bf16* WoT = (__bf16*)p; p += (size_t)MQA_D * MQA_D * 2;      // 2 MB
  __bf16* Qbf = (__bf16*)p; p += (size_t)MQA_M * MQA_D * 2;      // 8 MB
  __bf16* Kbf = (__bf16*)p; p += (size_t)MQA_M * MQA_HD * 2;     // 512 KB
  __bf16* Vt  = (__bf16*)p; p += (size_t)MQA_HD * MQA_M * 2;     // 512 KB
  __bf16* Ctx = (__bf16*)p; p += (size_t)MQA_M * MQA_D * 2;      // 8 MB

  // 0) precision / layout prep
  {
    const int n = MQA_M * MQA_D;
    cvt_bf16_kernel<<<(n + 255) / 256, 256, 0, stream>>>(Qin, Xbf, n);
  }
  transpose_cvt_kernel<<<(MQA_D * MQA_D + 255) / 256, 256, 0, stream>>>(
      Wq, WqT, MQA_D, MQA_D);
  transpose_cvt_kernel<<<(MQA_D * MQA_HD + 255) / 256, 256, 0, stream>>>(
      Wk, WkT, MQA_D, MQA_HD);
  transpose_cvt_kernel<<<(MQA_D * MQA_HD + 255) / 256, 256, 0, stream>>>(
      Wv, WvT, MQA_D, MQA_HD);
  transpose_cvt_kernel<<<(MQA_D * MQA_D + 255) / 256, 256, 0, stream>>>(
      Wo, WoT, MQA_D, MQA_D);

  // 1) fused QKV projection (WMMA, 32x64 per wave)
  qkv_gemm_kernel<<<dim3(MQA_M / 32, 16 + 1 + 1), 32, 0, stream>>>(
      Xbf, WqT, WkT, WvT, Qbf, Kbf, Vt);

  // 2) flash MQA attention (WMMA + async K/V staging to LDS)
  mqa_flash_kernel<<<dim3(MQA_S / 64, MQA_H, MQA_B), 128, 0, stream>>>(
      Qbf, Kbf, Vt, Ctx);

  // 3) output projection (WMMA, fp32 out)
  out_gemm_kernel<<<dim3(MQA_M / 32, MQA_D / 64), 32, 0, stream>>>(Ctx, WoT, Out);
}
